// NTXentLoss_52312701666200
// MI455X (gfx1250) — matrix operand order repaired and checked by reading, not compiled, not run on hardware
//
#include <hip/hip_runtime.h>
#include <math.h>

typedef __attribute__((ext_vector_type(16))) _Float16 v16h;
typedef __attribute__((ext_vector_type(8)))  _Float16 v8h;
typedef __attribute__((ext_vector_type(8)))  float    v8f;

#define N_ROWS   4096
#define DIM      256
#define TWO_N    8192
#define N_TILES  (TWO_N / 16)   /* 512 column tiles */
#define TEMP_INV 2.0f           /* 1 / TEMPERATURE(0.5) */
#define EPS      1e-8f

// ---------------------------------------------------------------------------
// Kernel 0: zero the scalar output (graph-replay deterministic init)
// ---------------------------------------------------------------------------
__global__ void ntxent_zero_kernel(float* out) {
    if (threadIdx.x == 0) out[0] = 0.0f;
}

// ---------------------------------------------------------------------------
// Kernel 1: row-normalize concat(z_i, z_j) -> f16 rows in workspace.
// One wave (32 lanes) per row; 8 rows per 256-thread block.
// ---------------------------------------------------------------------------
__global__ __launch_bounds__(256)
void ntxent_normalize_kernel(const float* __restrict__ zi,
                             const float* __restrict__ zj,
                             _Float16* __restrict__ zn) {
    const int wave = threadIdx.x >> 5;
    const int lane = threadIdx.x & 31;
    const int row  = blockIdx.x * 8 + wave;           // 0 .. 8191

    const float* src = (row < N_ROWS) ? (zi + (size_t)row * DIM)
                                      : (zj + (size_t)(row - N_ROWS) * DIM);
    float vals[8];
    float ss = 0.0f;
#pragma unroll
    for (int i = 0; i < 8; ++i) {
        float v = src[lane + 32 * i];
        vals[i] = v;
        ss += v * v;
    }
#pragma unroll
    for (int m = 16; m >= 1; m >>= 1) ss += __shfl_xor(ss, m, 32);

    float nrm = fmaxf(sqrtf(ss), EPS);
    float inv = 1.0f / nrm;

    _Float16* dst = zn + (size_t)row * DIM;
#pragma unroll
    for (int i = 0; i < 8; ++i)
        dst[lane + 32 * i] = (_Float16)(vals[i] * inv);
}

// ---------------------------------------------------------------------------
// Helpers for the fused Gram/NT-Xent kernel.
// ---------------------------------------------------------------------------

// B fragment layout (16-bit B 32x16): lane<16 holds K = kb*32 + {0..7, 16..23} of
// column N=lane; lane>=16 holds K = kb*32 + {8..15, 24..31} of column N=lane-16.
// Two 16B chunks per lane per kb.
__device__ __forceinline__ void load_b_tile(v16h b[8],
                                            const _Float16* __restrict__ zn,
                                            int colTile, int l15, int half) {
    const _Float16* bRow = zn + (size_t)(colTile * 16 + l15) * DIM;
#pragma unroll
    for (int kb = 0; kb < 8; ++kb) {
        const int k0 = kb * 32 + half * 8;
        v8h lo = *(const v8h*)(bRow + k0);
        v8h hi = *(const v8h*)(bRow + k0 + 16);
        b[kb] = __builtin_shufflevector(lo, hi,
            0, 1, 2, 3, 4, 5, 6, 7, 8, 9, 10, 11, 12, 13, 14, 15);
    }
}

// 8x v_wmma_f32_16x16x32_f16 (K = 256) + fused softmax-denominator epilogue.
// C/D layout: VGPR r, lanes 0-15 -> M = r; lanes 16-31 -> M = r + 8.
__device__ __forceinline__ void compute_tile(const v16h a[8], const v16h b[8],
                                             int rowBase, int colTile,
                                             int l15, int half,
                                             float sumE[8], float posS[8]) {
    v8f acc = {};
#pragma unroll
    for (int kb = 0; kb < 8; ++kb) {
        acc = __builtin_amdgcn_wmma_f32_16x16x32_f16(
            /*neg_a=*/false, a[kb], /*neg_b=*/false, b[kb],
            /*c_mod=*/(short)0, acc, /*reuse_a=*/false, /*reuse_b=*/false);
    }
    const int col = colTile * 16 + l15;
#pragma unroll
    for (int r = 0; r < 8; ++r) {
        const int row = rowBase + r + 8 * half;
        float s = acc[r] * TEMP_INV;           // sim / T
        float e = __expf(s);
        if (row == col) e = 0.0f;              // exclude self-similarity
        sumE[r] += e;
        const int pcol = (row < N_ROWS) ? row + N_ROWS : row - N_ROWS;
        if (col == pcol) posS[r] = s;          // positive-pair logit
    }
}

// ---------------------------------------------------------------------------
// Kernel 2: fused Gram-matrix + NT-Xent. One wave per 16-row tile; ping-pong
// double-buffered column tiles so global loads overlap the WMMA chain.
// ---------------------------------------------------------------------------
__global__ __launch_bounds__(128)
void ntxent_main_kernel(const _Float16* __restrict__ zn,
                        float* __restrict__ out) {
    const int waveInBlock = threadIdx.x >> 5;
    const int lane        = threadIdx.x & 31;
    const int rowTile     = blockIdx.x * 4 + waveInBlock;   // 0 .. 511
    const int rowBase     = rowTile * 16;
    const int half        = lane >> 4;                      // 0 or 1
    const int l15         = lane & 15;

    // A fragments: rows rowBase + (lane&15), all of K = 0..255, held in regs.
    // 16-bit A 16x32 layout: lane<16 holds K = kb*32 + {0..7, 16..23},
    // lane>=16 holds K = kb*32 + {8..15, 24..31}  -> two 16B chunks.
    const _Float16* aRow = zn + (size_t)(rowBase + l15) * DIM;
    v16h aFrag[8];
#pragma unroll
    for (int kb = 0; kb < 8; ++kb) {
        const int k0 = kb * 32 + half * 8;
        v8h lo = *(const v8h*)(aRow + k0);
        v8h hi = *(const v8h*)(aRow + k0 + 16);
        aFrag[kb] = __builtin_shufflevector(lo, hi,
            0, 1, 2, 3, 4, 5, 6, 7, 8, 9, 10, 11, 12, 13, 14, 15);
    }

    float sumE[8];
    float posS[8];
#pragma unroll
    for (int r = 0; r < 8; ++r) { sumE[r] = 0.0f; posS[r] = 0.0f; }

    // Ping-pong double buffer over column tiles (N_TILES = 512, even).
    v16h buf0[8], buf1[8];
    load_b_tile(buf0, zn, 0, l15, half);

    for (int ct = 0; ct < N_TILES; ct += 2) {
        // Issue loads for tile ct+1, then compute tile ct (no wait inside).
        __builtin_prefetch(zn + (size_t)((ct + 1) * 16 + l15) * DIM + 16 * DIM, 0, 1);
        load_b_tile(buf1, zn, ct + 1, l15, half);
        compute_tile(aFrag, buf0, rowBase, ct, l15, half, sumE, posS);

        // Issue loads for tile ct+2 (address-clamped tail), compute tile ct+1.
        const int ctNext = (ct + 2 < N_TILES) ? (ct + 2) : (N_TILES - 1);
        load_b_tile(buf0, zn, ctNext, l15, half);
        compute_tile(aFrag, buf1, rowBase, ct + 1, l15, half, sumE, posS);
    }

    // Reduce each row across its 16-lane group; combine halves; accumulate.
    float waveLoss = 0.0f;
#pragma unroll
    for (int r = 0; r < 8; ++r) {
        float v = sumE[r];
        float p = posS[r];
#pragma unroll
        for (int m = 8; m >= 1; m >>= 1) {
            v += __shfl_xor(v, m, 32);
            p += __shfl_xor(p, m, 32);
        }
        // loss_row = -log(exp(pos/T)/denom) = log(denom) - pos/T
        waveLoss += __logf(v) - p;
    }
    const float other = __shfl_xor(waveLoss, 16, 32);  // lane0 <-> lane16
    if (lane == 0)
        atomicAdd(out, (waveLoss + other) * (1.0f / (float)TWO_N));
}

// ---------------------------------------------------------------------------
extern "C" void kernel_launch(void* const* d_in, const int* in_sizes, int n_in,
                              void* d_out, int out_size, void* d_ws, size_t ws_size,
                              hipStream_t stream) {
    const float* zi  = (const float*)d_in[0];
    const float* zj  = (const float*)d_in[1];
    float*       out = (float*)d_out;
    _Float16*    zn  = (_Float16*)d_ws;   // needs 8192*256*2 = 4 MB

    hipLaunchKernelGGL(ntxent_zero_kernel, dim3(1), dim3(32), 0, stream, out);
    hipLaunchKernelGGL(ntxent_normalize_kernel, dim3(TWO_N / 8), dim3(256), 0, stream,
                       zi, zj, zn);
    hipLaunchKernelGGL(ntxent_main_kernel, dim3(N_TILES / 4), dim3(128), 0, stream,
                       zn, out);
}
